// Encoder_20005957665082
// MI455X (gfx1250) — compile-verified, matrix operand order checked by reading, not run on hardware
//
#include <hip/hip_runtime.h>

typedef _Float16 f16;
typedef __attribute__((ext_vector_type(16))) _Float16 v16h;
typedef __attribute__((ext_vector_type(8)))  _Float16 v8h;
typedef __attribute__((ext_vector_type(8)))  float    v8f;
typedef __attribute__((ext_vector_type(4)))  int      v4i;

#define CB  32      // batch
#define CT  512     // time
#define CE  512     // embed dim
#define CH  256     // lstm hidden
#define CF  512     // conv filters
#define CK  5       // conv width
#define G4H 1024    // 4*H

// ---- gfx1250 async global->LDS copy (ASYNCcnt), guarded by __has_builtin ----
#if defined(__has_builtin)
#  if __has_builtin(__builtin_amdgcn_global_load_async_to_lds_b128) && \
      __has_builtin(__builtin_amdgcn_s_wait_asynccnt)
#    define USE_ASYNC_LDS 1
#  endif
#endif
#ifndef USE_ASYNC_LDS
#  define USE_ASYNC_LDS 0
#endif

#if USE_ASYNC_LDS
// builtin signature (from clang diagnostic): (int4 AS1*, int4 AS3*, imm, imm)
typedef __attribute__((address_space(1))) v4i* gv4i_t;
typedef __attribute__((address_space(3))) v4i* lv4i_t;
__device__ __forceinline__ void async_cp16(const f16* g, f16* l) {
  __builtin_amdgcn_global_load_async_to_lds_b128(
      (gv4i_t)(v4i*)(void*)g, (lv4i_t)(v4i*)(void*)l, 0, 0);
}
__device__ __forceinline__ void async_wait0() {
  __builtin_amdgcn_s_wait_asynccnt(0);
}
#endif

// ---------------------------------------------------------------------------
// WMMA fragment loader. Row-major source (f16, leading dim `ld` elements).
// CDNA5 16-bit A/B layout (wave32): lanes 0-15 -> row m=lane, K 0..7 & 16..23;
// lanes 16-31 -> row m=lane-16, K 8..15 & 24..31. Two 16B loads per lane.
// ---------------------------------------------------------------------------
__device__ __forceinline__ v16h load_frag(const f16* base, int ld) {
  const int lane = threadIdx.x & 31;
  const int m    = lane & 15;
  const int kh   = (lane >> 4) << 3;          // 0 or 8
  const f16* p = base + m * ld + kh;
  v8h lo = *(const v8h*)(p);
  v8h hi = *(const v8h*)(p + 16);
  v16h f;
#pragma unroll
  for (int i = 0; i < 8; ++i) { f[i] = lo[i]; f[i + 8] = hi[i]; }
  return f;
}

__device__ __forceinline__ v8f wmma_f16(v16h a, v16h b, v8f c) {
  return __builtin_amdgcn_wmma_f32_16x16x32_f16(false, a, false, b,
                                                (short)0, c, false, false);
}

__device__ __forceinline__ float sigmoid_f(float x) {
  return 1.0f / (1.0f + __expf(-x));
}
__device__ __forceinline__ float tanh_f(float x) {
  x = fminf(fmaxf(x, -15.0f), 15.0f);
  float e = __expf(2.0f * x);
  return (e - 1.0f) / (e + 1.0f);
}

// ---------------------------------------------------------------------------
// Weight / BN preparation (cast to f16, transpose to [N][K] row-major)
// ---------------------------------------------------------------------------
__global__ void k_prep_convw(const float* __restrict__ w, f16* __restrict__ wt) {
  // src: [3][5][Cin][Cout]  ->  dst: [3][5][Cout][Cin]
  size_t idx = (size_t)blockIdx.x * blockDim.x + threadIdx.x;
  if (idx >= (size_t)3 * CK * CF * CF) return;
  int ci = (int)(idx & (CF - 1));
  size_t r = idx >> 9;
  int co = (int)(r & (CF - 1));
  size_t lk = r >> 9;
  wt[idx] = (f16)w[(lk * CF + ci) * CF + co];
}

__global__ void k_prep_wx(const float* __restrict__ w, f16* __restrict__ wt) {
  // src: [2][F][4H] -> dst: [2][4H][F]
  size_t idx = (size_t)blockIdx.x * blockDim.x + threadIdx.x;
  if (idx >= (size_t)2 * CF * G4H) return;
  int f = (int)(idx & (CF - 1));
  size_t r = idx >> 9;
  int g = (int)(r & (G4H - 1));
  int d = (int)(r >> 10);
  wt[idx] = (f16)w[((size_t)d * CF + f) * G4H + g];
}

__global__ void k_prep_wh(const float* __restrict__ w, f16* __restrict__ wt) {
  // src: [2][H][4H] -> dst: [2][4H][H]
  size_t idx = (size_t)blockIdx.x * blockDim.x + threadIdx.x;
  if (idx >= (size_t)2 * CH * G4H) return;
  int h = (int)(idx & (CH - 1));
  size_t r = idx >> 8;
  int g = (int)(r & (G4H - 1));
  int d = (int)(r >> 10);
  wt[idx] = (f16)w[((size_t)d * CH + h) * G4H + g];
}

__global__ void k_prep_bn(const float* __restrict__ g, const float* __restrict__ be,
                          const float* __restrict__ mu, const float* __restrict__ var,
                          const float* __restrict__ cb,
                          float* __restrict__ inv, float* __restrict__ sh) {
  int i = blockIdx.x * blockDim.x + threadIdx.x;
  if (i >= 3 * CF) return;
  float iv = g[i] * rsqrtf(var[i] + 1e-3f);
  inv[i] = iv;
  sh[i]  = (cb[i] - mu[i]) * iv + be[i];   // y = acc*inv + shift (conv bias folded)
}

__global__ void k_embed(const int* __restrict__ tokens, const float* __restrict__ emb,
                        f16* __restrict__ X) {
  int row = blockIdx.x;                      // b*T + t
  int tok = tokens[row];
  const float* s = emb + (size_t)tok * CE;
  f16* d = X + (size_t)row * CE;
  for (int i = threadIdx.x; i < CE; i += blockDim.x) d[i] = (f16)s[i];
}

// ---------------------------------------------------------------------------
// Conv1d(K=5, SAME) + BN + ReLU as a 5-tap accumulated GEMM.
// Block = 128 threads (4 waves); tile = 128 rows x 64 channels.
// Double-buffered LDS pipeline: stage chunk i+1 (async global->LDS) while
// computing chunk i; one barrier per chunk. 80 chunks = 5 taps x 16 k-chunks.
// ---------------------------------------------------------------------------
__global__ void __launch_bounds__(128)
k_conv_bn_relu(const f16* __restrict__ Xin, const f16* __restrict__ WT,
               const float* __restrict__ inv, const float* __restrict__ shift,
               f16* __restrict__ Xout)
{
  __shared__ f16 As[2][128][40];   // stride 40 halves = 80B, spreads banks
  __shared__ f16 Bs[2][64][40];
  const int tid  = threadIdx.x;
  const int wave = tid >> 5;
  const int lane = tid & 31;
  const int m0 = blockIdx.x * 128;           // row tile (row = b*T + t)
  const int n0 = blockIdx.y * 64;            // out-channel tile
  const int t0 = m0 & (CT - 1);
  const int brow = tid >> 1;                 // 0..63 : staged B row
  const int bseg = (tid & 1) << 4;           // 0 or 16 halves

  v8f acc[2][4] = {};

  // pointer/validity computation for flattened iteration (tap, kc)
  const f16* ap; const f16* bp; bool aok;
  auto ptrs = [&](int it) {
    const int tap = it >> 4;
    const int kc  = (it & 15) << 5;
    const int dt  = tap - 2;
    aok = (unsigned)(t0 + tid + dt) < (unsigned)CT;
    ap  = Xin + (size_t)(m0 + tid + dt) * CF + kc;
    bp  = WT + (size_t)tap * CF * CF + (size_t)(n0 + brow) * CF + kc + bseg;
  };

#if USE_ASYNC_LDS
  auto stage = [&](int it, int buf) {
    ptrs(it);
    f16* adst = &As[buf][tid][0];
    if (aok) {
      async_cp16(ap +  0, adst +  0);
      async_cp16(ap +  8, adst +  8);
      async_cp16(ap + 16, adst + 16);
      async_cp16(ap + 24, adst + 24);
    } else {
      v8h z = {0, 0, 0, 0, 0, 0, 0, 0};
      *(v8h*)(adst +  0) = z; *(v8h*)(adst +  8) = z;
      *(v8h*)(adst + 16) = z; *(v8h*)(adst + 24) = z;
    }
    f16* bdst = &Bs[buf][brow][bseg];
    async_cp16(bp + 0, bdst + 0);
    async_cp16(bp + 8, bdst + 8);
  };
#else
  v8h ra[4], rb[2];
  auto stage_load = [&](int it) {
    ptrs(it);
    if (aok) {
      ra[0] = *(const v8h*)(ap +  0); ra[1] = *(const v8h*)(ap +  8);
      ra[2] = *(const v8h*)(ap + 16); ra[3] = *(const v8h*)(ap + 24);
    } else {
      v8h z = {0, 0, 0, 0, 0, 0, 0, 0};
      ra[0] = z; ra[1] = z; ra[2] = z; ra[3] = z;
    }
    rb[0] = *(const v8h*)(bp + 0);
    rb[1] = *(const v8h*)(bp + 8);
    if (aok) __builtin_prefetch(ap + 64, 0, 1);
    __builtin_prefetch(bp + 64, 0, 1);
  };
  auto stage_store = [&](int buf) {
    f16* adst = &As[buf][tid][0];
    *(v8h*)(adst +  0) = ra[0]; *(v8h*)(adst +  8) = ra[1];
    *(v8h*)(adst + 16) = ra[2]; *(v8h*)(adst + 24) = ra[3];
    f16* bdst = &Bs[buf][brow][bseg];
    *(v8h*)(bdst + 0) = rb[0];
    *(v8h*)(bdst + 8) = rb[1];
  };
#endif

  const int NIT = CK * (CF / 32);   // 80

  // prologue: stage chunk 0 into buffer 0
#if USE_ASYNC_LDS
  stage(0, 0);
  async_wait0();
#else
  stage_load(0);
  stage_store(0);
#endif
  __syncthreads();

  for (int it = 0; it < NIT; ++it) {
    const int buf = it & 1;
#if USE_ASYNC_LDS
    if (it + 1 < NIT) stage(it + 1, buf ^ 1);
#else
    if (it + 1 < NIT) stage_load(it + 1);
#endif
    // compute chunk `it` from LDS buffer `buf`
    v16h a0 = load_frag(&As[buf][wave * 32][0],      40);
    v16h a1 = load_frag(&As[buf][wave * 32 + 16][0], 40);
#pragma unroll
    for (int nt = 0; nt < 4; ++nt) {
      v16h bf = load_frag(&Bs[buf][nt * 16][0], 40);
      acc[0][nt] = wmma_f16(a0, bf, acc[0][nt]);
      acc[1][nt] = wmma_f16(a1, bf, acc[1][nt]);
    }
#if USE_ASYNC_LDS
    if (it + 1 < NIT) async_wait0();
#else
    if (it + 1 < NIT) stage_store(buf ^ 1);
#endif
    __syncthreads();
  }

  // Epilogue: BN + ReLU, store f16. D layout: VGPR j -> M=j(+8 hi lanes), N=lane&15.
  const int mbase = (lane >> 4) << 3;
#pragma unroll
  for (int mt = 0; mt < 2; ++mt) {
    const int mrow = m0 + wave * 32 + mt * 16 + mbase;
#pragma unroll
    for (int nt = 0; nt < 4; ++nt) {
      const int n = n0 + nt * 16 + (lane & 15);
      const float iv = inv[n], sh = shift[n];
#pragma unroll
      for (int j = 0; j < 8; ++j) {
        float y = acc[mt][nt][j] * iv + sh;
        y = y > 0.0f ? y : 0.0f;
        Xout[(size_t)(mrow + j) * CF + n] = (f16)y;
      }
    }
  }
}

// ---------------------------------------------------------------------------
// LSTM input projections: XW[dir][t*B+b][g] = x[b][t] . Wx[dir] + bias  (f16)
// Same pipelined 128x64 tile; A rows permuted (r = t*32+b -> x row b*512+t).
// ---------------------------------------------------------------------------
__global__ void __launch_bounds__(128)
k_xw_gemm(const f16* __restrict__ X, const f16* __restrict__ WxT,
          const float* __restrict__ bias, f16* __restrict__ XW)
{
  __shared__ f16 As[2][128][40];
  __shared__ f16 Bs[2][64][40];
  const int tid  = threadIdx.x;
  const int wave = tid >> 5;
  const int lane = tid & 31;
  const int dir = blockIdx.z;
  const int m0 = blockIdx.x * 128;
  const int n0 = blockIdx.y * 64;
  const int brow = tid >> 1;
  const int bseg = (tid & 1) << 4;

  const int r = m0 + tid;
  const int t = r >> 5;
  const int b = r & (CB - 1);
  const f16* abase = X + ((size_t)b * CT + t) * CF;
  const f16* bbase = WxT + ((size_t)dir * G4H + n0 + brow) * CF + bseg;

  v8f acc[2][4] = {};

#if USE_ASYNC_LDS
  auto stage = [&](int it, int buf) {
    const f16* ap = abase + (it << 5);
    const f16* bp = bbase + (it << 5);
    f16* adst = &As[buf][tid][0];
    async_cp16(ap +  0, adst +  0);
    async_cp16(ap +  8, adst +  8);
    async_cp16(ap + 16, adst + 16);
    async_cp16(ap + 24, adst + 24);
    f16* bdst = &Bs[buf][brow][bseg];
    async_cp16(bp + 0, bdst + 0);
    async_cp16(bp + 8, bdst + 8);
  };
#else
  v8h ra[4], rb[2];
  auto stage_load = [&](int it) {
    const f16* ap = abase + (it << 5);
    const f16* bp = bbase + (it << 5);
    ra[0] = *(const v8h*)(ap +  0); ra[1] = *(const v8h*)(ap +  8);
    ra[2] = *(const v8h*)(ap + 16); ra[3] = *(const v8h*)(ap + 24);
    rb[0] = *(const v8h*)(bp + 0);
    rb[1] = *(const v8h*)(bp + 8);
    __builtin_prefetch(ap + 64, 0, 1);
    __builtin_prefetch(bp + 64, 0, 1);
  };
  auto stage_store = [&](int buf) {
    f16* adst = &As[buf][tid][0];
    *(v8h*)(adst +  0) = ra[0]; *(v8h*)(adst +  8) = ra[1];
    *(v8h*)(adst + 16) = ra[2]; *(v8h*)(adst + 24) = ra[3];
    f16* bdst = &Bs[buf][brow][bseg];
    *(v8h*)(bdst + 0) = rb[0];
    *(v8h*)(bdst + 8) = rb[1];
  };
#endif

  const int NIT = CF / 32;   // 16

#if USE_ASYNC_LDS
  stage(0, 0);
  async_wait0();
#else
  stage_load(0);
  stage_store(0);
#endif
  __syncthreads();

  for (int it = 0; it < NIT; ++it) {
    const int buf = it & 1;
#if USE_ASYNC_LDS
    if (it + 1 < NIT) stage(it + 1, buf ^ 1);
#else
    if (it + 1 < NIT) stage_load(it + 1);
#endif
    v16h a0 = load_frag(&As[buf][wave * 32][0],      40);
    v16h a1 = load_frag(&As[buf][wave * 32 + 16][0], 40);
#pragma unroll
    for (int nt = 0; nt < 4; ++nt) {
      v16h bf = load_frag(&Bs[buf][nt * 16][0], 40);
      acc[0][nt] = wmma_f16(a0, bf, acc[0][nt]);
      acc[1][nt] = wmma_f16(a1, bf, acc[1][nt]);
    }
#if USE_ASYNC_LDS
    if (it + 1 < NIT) async_wait0();
#else
    if (it + 1 < NIT) stage_store(buf ^ 1);
#endif
    __syncthreads();
  }

  const int mbase = (lane >> 4) << 3;
  f16* XWd = XW + (size_t)dir * (CT * CB) * G4H;
#pragma unroll
  for (int mt = 0; mt < 2; ++mt) {
    const int mrow = m0 + wave * 32 + mt * 16 + mbase;
#pragma unroll
    for (int nt = 0; nt < 4; ++nt) {
      const int n = n0 + nt * 16 + (lane & 15);
      const float bb = bias[dir * G4H + n];
#pragma unroll
      for (int j = 0; j < 8; ++j)
        XWd[(size_t)(mrow + j) * G4H + n] = (f16)(acc[mt][nt][j] + bb);
    }
  }
}

// ---------------------------------------------------------------------------
// Recurrent scan with zoneout. One persistent workgroup per direction,
// 512 threads = 16 waves. Wave w owns gate columns {g*256 + w*16 .. +15} for
// all four gates g, so i/f/g/o for the same (b,h) sit in the same lane and
// accumulator slot -> elementwise phase runs out of registers (no gate LDS).
// h (f16) and c (f32) state live in LDS across all 512 steps.
// ---------------------------------------------------------------------------
__global__ void __launch_bounds__(512)
k_lstm_scan(const f16* __restrict__ WhT, const f16* __restrict__ XW,
            float* __restrict__ out)
{
  __shared__ f16  hbuf[CB][CH + 8];     // 32 x 264 f16
  __shared__ float cbuf[CB][CH];        // 32 x 256 f32
  const int tid  = threadIdx.x;
  const int wave = tid >> 5;            // 0..15
  const int lane = tid & 31;
  const int dir  = blockIdx.x;
  const f16* Wd  = WhT + (size_t)dir * G4H * CH;        // [1024][256] (N x K)
  const f16* XWd = XW + (size_t)dir * (CT * CB) * G4H;

  for (int i = tid; i < CB * (CH + 8); i += 512) ((f16*)hbuf)[i] = (f16)0.0f;
  for (int i = tid; i < CB * CH; i += 512) ((float*)cbuf)[i] = 0.0f;
  __syncthreads();

  const int hh    = wave * 16 + (lane & 15);  // this lane's h column
  const int mbase = (lane >> 4) << 3;

  for (int step = 0; step < CT; ++step) {
    const int t = dir ? (CT - 1 - step) : step;

    // gates[b][g*256+hh] = h . WhT   (M=32 in two 16-row tiles, K=256)
    v8f acc[2][4] = {};
#pragma unroll 2
    for (int kc = 0; kc < CH; kc += 32) {
      v16h a0 = load_frag(&hbuf[0][kc],  CH + 8);
      v16h a1 = load_frag(&hbuf[16][kc], CH + 8);
#pragma unroll
      for (int gt = 0; gt < 4; ++gt) {
        const f16* bb = Wd + (size_t)(gt * CH + wave * 16) * CH + kc;  // L2-resident
        v16h bf = load_frag(bb, CH);
        acc[0][gt] = wmma_f16(a0, bf, acc[0][gt]);
        acc[1][gt] = wmma_f16(a1, bf, acc[1][gt]);
      }
    }
    __syncthreads();   // everyone done reading hbuf before it is overwritten

    const f16* xwrow = XWd + (size_t)t * CB * G4H;
#pragma unroll
    for (int mt = 0; mt < 2; ++mt) {
#pragma unroll
      for (int j = 0; j < 8; ++j) {
        const int b = mt * 16 + mbase + j;
        const size_t xb = (size_t)b * G4H;
        float gi = acc[mt][0][j] + (float)xwrow[xb + 0 * CH + hh];
        float gf = acc[mt][1][j] + (float)xwrow[xb + 1 * CH + hh];
        float gg = acc[mt][2][j] + (float)xwrow[xb + 2 * CH + hh];
        float go = acc[mt][3][j] + (float)xwrow[xb + 3 * CH + hh];
        float i_ = sigmoid_f(gi);
        float f_ = sigmoid_f(gf);
        float o_ = sigmoid_f(go);
        float g_ = tanh_f(gg);
        float c_old = cbuf[b][hh];
        float h_old = (float)hbuf[b][hh];
        float cn = f_ * c_old + i_ * g_;
        float hn = o_ * tanh_f(cn);
        out[((size_t)b * CT + t) * (2 * CH) + dir * CH + hh] = hn;  // pre-blend h
        cbuf[b][hh] = 0.9f * cn + 0.1f * c_old;                     // zoneout blend
        hbuf[b][hh] = (f16)(0.9f * hn + 0.1f * h_old);
      }
    }
    __syncthreads();
  }
}

// ---------------------------------------------------------------------------
extern "C" void kernel_launch(void* const* d_in, const int* in_sizes, int n_in,
                              void* d_out, int out_size, void* d_ws, size_t ws_size,
                              hipStream_t stream) {
  (void)in_sizes; (void)n_in; (void)out_size; (void)ws_size;
  const int*   tokens = (const int*)d_in[0];
  const float* emb    = (const float*)d_in[1];
  const float* conv_w = (const float*)d_in[2];
  const float* conv_b = (const float*)d_in[3];
  const float* bn_g   = (const float*)d_in[4];
  const float* bn_b   = (const float*)d_in[5];
  const float* bn_m   = (const float*)d_in[6];
  const float* bn_v   = (const float*)d_in[7];
  const float* l_wx   = (const float*)d_in[8];
  const float* l_wh   = (const float*)d_in[9];
  const float* l_b    = (const float*)d_in[10];
  float* out = (float*)d_out;

  char* p = (char*)d_ws;
  auto take = [&](size_t bytes) -> char* {
    char* q = p; p += (bytes + 255) & ~(size_t)255; return q;
  };
  f16*   Xa    = (f16*)take((size_t)CB * CT * CF * 2);       // 16 MB
  f16*   Xb    = (f16*)take((size_t)CB * CT * CF * 2);       // 16 MB
  f16*   WcT   = (f16*)take((size_t)3 * CK * CF * CF * 2);   // 7.5 MB
  f16*   WxT   = (f16*)take((size_t)2 * G4H * CF * 2);       // 2 MB
  f16*   WhT   = (f16*)take((size_t)2 * G4H * CH * 2);       // 1 MB
  float* bninv = (float*)take((size_t)3 * CF * 4);
  float* bnsh  = (float*)take((size_t)3 * CF * 4);
  f16*   XW    = (f16*)take((size_t)2 * CT * CB * G4H * 2);  // 64 MB

  // weight/BN prep
  k_prep_convw<<<(3 * CK * CF * CF) / 256, 256, 0, stream>>>(conv_w, WcT);
  k_prep_wx  <<<(2 * CF * G4H) / 256, 256, 0, stream>>>(l_wx, WxT);
  k_prep_wh  <<<(2 * CH * G4H) / 256, 256, 0, stream>>>(l_wh, WhT);
  k_prep_bn  <<<(3 * CF + 255) / 256, 256, 0, stream>>>(bn_g, bn_b, bn_m, bn_v,
                                                        conv_b, bninv, bnsh);
  // embedding gather -> f16
  k_embed<<<CB * CT, 128, 0, stream>>>(tokens, emb, Xa);

  // 3 conv+BN+ReLU layers (ping-pong)
  dim3 cg(CB * CT / 128, CF / 64);
  k_conv_bn_relu<<<cg, 128, 0, stream>>>(Xa, WcT + (size_t)0 * CK * CF * CF,
                                         bninv + 0 * CF, bnsh + 0 * CF, Xb);
  k_conv_bn_relu<<<cg, 128, 0, stream>>>(Xb, WcT + (size_t)1 * CK * CF * CF,
                                         bninv + 1 * CF, bnsh + 1 * CF, Xa);
  k_conv_bn_relu<<<cg, 128, 0, stream>>>(Xa, WcT + (size_t)2 * CK * CF * CF,
                                         bninv + 2 * CF, bnsh + 2 * CF, Xb);

  // LSTM input projections for both directions
  dim3 xg(CB * CT / 128, G4H / 64, 2);
  k_xw_gemm<<<xg, 128, 0, stream>>>(Xb, WxT, l_b, XW);

  // recurrent scan (one persistent workgroup per direction)
  k_lstm_scan<<<2, 512, 0, stream>>>(WhT, XW, out);
}
